// ContrastiveLossELI5_43207370998050
// MI455X (gfx1250) — matrix operand order repaired and checked by reading, not compiled, not run on hardware
//
#include <hip/hip_runtime.h>
#include <stdint.h>

typedef __attribute__((ext_vector_type(16))) _Float16 v16h;
typedef __attribute__((ext_vector_type(8)))  _Float16 v8h;
typedef __attribute__((ext_vector_type(8)))  float    v8f;

#define N_ROWS 8192      // 2N
#define HALF_N 4096      // N
#define DIM    256
#define INV_T  2.0f      // 1 / temperature

#define SEGS           4                     // column segments (grid.y)
#define SEG_COLS       (N_ROWS / SEGS)       // 2048
#define CHUNK          64                    // columns staged per pipeline step
#define CHUNKS_PER_SEG (SEG_COLS / CHUNK)    // 32
#define LDS_PITCH      264                   // halves/row: 132 dwords, 4-bank skew

// ---------------------------------------------------------------------------
// Kernel 1: L2-normalize rows of concat(emb_i, emb_j) -> f16 reps in d_ws.
// One wave32 per row. Also zeroes the per-row denominator array and d_out.
// ---------------------------------------------------------------------------
__global__ __launch_bounds__(256) void nt_normalize(const float* __restrict__ emb_i,
                                                    const float* __restrict__ emb_j,
                                                    _Float16* __restrict__ reps,
                                                    float* __restrict__ denom,
                                                    float* __restrict__ out) {
  if (blockIdx.x == 0 && threadIdx.x == 0) out[0] = 0.0f;
  const int gid  = blockIdx.x * 256 + threadIdx.x;
  const int row  = gid >> 5;
  const int lane = threadIdx.x & 31;
  if (row >= N_ROWS) return;
  if (lane == 0) denom[row] = 0.0f;

  const float* src = (row < HALF_N) ? (emb_i + (size_t)row * DIM)
                                    : (emb_j + (size_t)(row - HALF_N) * DIM);
  const float4* s4 = (const float4*)src;
  float4 x0 = s4[lane * 2 + 0];
  float4 x1 = s4[lane * 2 + 1];
  float ss = x0.x * x0.x + x0.y * x0.y + x0.z * x0.z + x0.w * x0.w
           + x1.x * x1.x + x1.y * x1.y + x1.z * x1.z + x1.w * x1.w;
#pragma unroll
  for (int m = 16; m >= 1; m >>= 1) ss += __shfl_xor(ss, m, 32);
  const float inv = 1.0f / fmaxf(sqrtf(ss), 1e-12f);

  v8h h;
  h[0] = (_Float16)(x0.x * inv); h[1] = (_Float16)(x0.y * inv);
  h[2] = (_Float16)(x0.z * inv); h[3] = (_Float16)(x0.w * inv);
  h[4] = (_Float16)(x1.x * inv); h[5] = (_Float16)(x1.y * inv);
  h[6] = (_Float16)(x1.z * inv); h[7] = (_Float16)(x1.w * inv);
  ((v8h*)(reps + (size_t)row * DIM))[lane] = h;
}

// ---------------------------------------------------------------------------
// Kernel 2: fused Gram / exp / partial row-sum via WMMA. grid = (64, SEGS).
// blockIdx.x selects 128 rows (8 waves x 16-row strips); blockIdx.y selects a
// 2048-column segment. Column chunks are double-buffered into LDS with
// GLOBAL_LOAD_ASYNC_TO_LDS_B128 (ASYNCcnt), shared by all 8 waves.
// ---------------------------------------------------------------------------
__global__ __launch_bounds__(256) void nt_xent_wmma(const _Float16* __restrict__ reps,
                                                    float* __restrict__ denom,
                                                    float* __restrict__ pos) {
  __shared__ _Float16 lbuf[2][CHUNK][LDS_PITCH];   // 2 x 33792 B = 67.5 KB

  const int tid  = threadIdx.x;
  const int lane = tid & 31;
  const int half = lane >> 4;
  const int lrow = lane & 15;
  // force wave-uniform scalars so tile classification is SALU, not EXEC games
  const int row_base = __builtin_amdgcn_readfirstlane(blockIdx.x * 128 + (tid >> 5) * 16);
  const int seg      = blockIdx.y;
  const int seg_col0 = seg * SEG_COLS;
  const int pos_base = row_base ^ HALF_N;          // tile holding positive pairs
  const bool pos_in_seg = (pos_base >> 11) == seg; // 2048 columns per segment

  // staging coordinates: 4 threads per staged row, 128 B (8 x b128) each
  const int sr = tid & 63;
  const int sq = tid >> 6;

  // Issue 8 async b128 copies global->LDS. INST_OFFSET is added to BOTH the
  // LDS and the global address (ISA 10.x / 15.18.3), so one address pair + 8
  // immediate offsets covers this thread's 128 B.
  auto stage = [&](int gchunk, int buf) {
    unsigned long long ga =
        (unsigned long long)(reps + ((size_t)(gchunk * CHUNK + sr) * DIM + sq * 64));
    unsigned int ldsa = (unsigned int)(uintptr_t)&lbuf[buf][sr][sq * 64];
#define A16(off)                                                            \
    asm volatile("global_load_async_to_lds_b128 %0, %1, off offset:" #off   \
                 :: "v"(ldsa), "v"(ga) : "memory")
    A16(0); A16(16); A16(32); A16(48); A16(64); A16(80); A16(96); A16(112);
#undef A16
  };

  // A fragments: this wave's 16 rows, all K = 256, contiguous k = 16*half + s per lane (B matches).
  v16h a[8];
  {
    const _Float16* rowp = reps + (size_t)(row_base + lrow) * DIM;
#pragma unroll
    for (int kk = 0; kk < 8; ++kk) {
      const int K0 = kk * 32;
      v8h lo = *(const v8h*)(rowp + K0 + half * 16);
      v8h hi = *(const v8h*)(rowp + K0 + half * 16 + 8);
      v16h av;
#pragma unroll
      for (int t = 0; t < 8; ++t) { av[t] = lo[t]; av[t + 8] = hi[t]; }
      a[kk] = av;
    }
  }

  float rsum[8], posv[8];
#pragma unroll
  for (int v = 0; v < 8; ++v) { rsum[v] = 0.0f; posv[v] = 0.0f; }

  // Prologue: stage this segment's first chunk into buffer 0.
  stage(seg * CHUNKS_PER_SEG, 0);
  asm volatile("s_wait_asynccnt 0x0" ::: "memory");
  __syncthreads();

  for (int c = 0; c < CHUNKS_PER_SEG; ++c) {
    const int cur = c & 1;
    if (c + 1 < CHUNKS_PER_SEG)
      stage(seg * CHUNKS_PER_SEG + c + 1, cur ^ 1);      // overlap fill

#pragma unroll
    for (int ct = 0; ct < CHUNK / 16; ++ct) {
      const int col_base = seg_col0 + c * CHUNK + ct * 16;
      const _Float16* colp = &lbuf[cur][ct * 16 + lrow][half * 16];
      v8f acc = {0.f, 0.f, 0.f, 0.f, 0.f, 0.f, 0.f, 0.f};
#pragma unroll
      for (int kk = 0; kk < 8; ++kk) {
        v16h b = *(const v16h*)(colp + kk * 32);
        acc = __builtin_amdgcn_wmma_f32_16x16x32_f16(false, a[kk], false, b,
                                                     (short)0, acc, false, false);
      }
      const bool diagTile = (col_base == row_base);      // scalar compare
      const bool posTile  = (col_base == pos_base);      // scalar compare
      if (diagTile || posTile) {                         // s_cbranch, 2/segment
#pragma unroll
        for (int v = 0; v < 8; ++v) {
          const bool mine  = (lrow == v + 8 * half);
          const float logit = acc[v] * INV_T;
          const float e     = __expf(logit);
          rsum[v] += (diagTile && mine) ? 0.0f : e;      // denom excludes diag
          posv[v] += (posTile  && mine) ? logit : 0.0f;
        }
      } else {
#pragma unroll
        for (int v = 0; v < 8; ++v) rsum[v] += __expf(acc[v] * INV_T);
      }
    }

    asm volatile("s_wait_asynccnt 0x0" ::: "memory");    // next chunk landed
    __syncthreads();                                     // visible block-wide
  }

  // Reduce across the 16 lanes of each half (xor 1,2,4,8 stays in-half).
#pragma unroll
  for (int v = 0; v < 8; ++v) {
    float s = rsum[v], p = posv[v];
#pragma unroll
    for (int m = 1; m <= 8; m <<= 1) {
      s += __shfl_xor(s, m, 32);
      p += __shfl_xor(p, m, 32);
    }
    rsum[v] = s; posv[v] = p;
  }

  if (lrow == 0) {   // lanes 0 (rows +0..7) and 16 (rows +8..15)
#pragma unroll
    for (int v = 0; v < 8; ++v)
      atomicAdd(&denom[row_base + 8 * half + v], rsum[v]);
    if (pos_in_seg) {                                    // single writer per row
#pragma unroll
      for (int v = 0; v < 8; ++v)
        pos[row_base + 8 * half + v] = posv[v];
    }
  }
}

// ---------------------------------------------------------------------------
// Kernel 3: loss = sum_i (log(denom_i) - pos_i) / (2N)
// ---------------------------------------------------------------------------
__global__ __launch_bounds__(256) void nt_finalize(const float* __restrict__ denom,
                                                   const float* __restrict__ pos,
                                                   float* __restrict__ out) {
  const int gid = blockIdx.x * 256 + threadIdx.x;        // 2048 threads
  float t = 0.0f;
#pragma unroll
  for (int i = 0; i < N_ROWS / 2048; ++i) {
    const int r = gid + i * 2048;
    t += __logf(denom[r]) - pos[r];
  }
#pragma unroll
  for (int m = 16; m >= 1; m >>= 1) t += __shfl_xor(t, m, 32);
  if ((threadIdx.x & 31) == 0)
    atomicAdd(out, t * (1.0f / (2.0f * (float)HALF_N)));
}

// ---------------------------------------------------------------------------
extern "C" void kernel_launch(void* const* d_in, const int* in_sizes, int n_in,
                              void* d_out, int out_size, void* d_ws, size_t ws_size,
                              hipStream_t stream) {
  const float* emb_i = (const float*)d_in[0];
  const float* emb_j = (const float*)d_in[1];
  float* out = (float*)d_out;

  _Float16* reps = (_Float16*)d_ws;                          // 4 MB
  float* denom = (float*)((char*)d_ws + (size_t)N_ROWS * DIM * 2);
  float* pos   = denom + N_ROWS;                             // +64 KB total

  nt_normalize<<<N_ROWS / 8, 256, 0, stream>>>(emb_i, emb_j, reps, denom, out);
  nt_xent_wmma<<<dim3(N_ROWS / 128, SEGS), 256, 0, stream>>>(reps, denom, pos);
  nt_finalize<<<2048 / 256, 256, 0, stream>>>(denom, pos, out);
}
